// CausalSelfAttention_2095944040722
// MI455X (gfx1250) — matrix operand order repaired and checked by reading, not compiled, not run on hardware
//
#include <hip/hip_runtime.h>
#include <hip/hip_bf16.h>

// Problem constants (from the reference)
#define B_  2
#define S_  2048
#define E_  1024
#define H_  16
#define HD_ 64
#define M_  (B_ * S_)   // 4096 flattened (b,s) rows

typedef __attribute__((ext_vector_type(16))) __bf16 bf16x16;
typedef __attribute__((ext_vector_type(8)))  __bf16 bf16x8;
typedef __attribute__((ext_vector_type(8)))  float  f32x8;

// Build a 16-element bf16 WMMA fragment from two contiguous 16-byte runs
// (callers use contiguous k = 16*half + s for both A and B: VGPR0-3 = first run, VGPR4-7 = second).
static __device__ inline bf16x16 frag16(const __bf16* p0, const __bf16* p1) {
  bf16x8 lo = *(const bf16x8*)p0;
  bf16x8 hi = *(const bf16x8*)p1;
  bf16x16 r;
#pragma unroll
  for (int i = 0; i < 8; ++i) { r[i] = lo[i]; r[i + 8] = hi[i]; }
  return r;
}

static __device__ inline f32x8 wmma_bf16(bf16x16 a, bf16x16 b, f32x8 c) {
  // D = A(16x32) * B(32x16) + C(16x16), f32 accumulate
  return __builtin_amdgcn_wmma_f32_16x16x32_bf16(false, a, false, b, (short)0, c,
                                                 false, false);
}

// ---------------------------------------------------------------------------
// fp32 -> bf16 elementwise convert (x activations)
// ---------------------------------------------------------------------------
__global__ void cvt_f32_bf16(const float* __restrict__ src,
                             __bf16* __restrict__ dst, int n) {
  int i = (blockIdx.x * blockDim.x + threadIdx.x) * 4;
  if (i + 3 < n) {
    const float4 v = *(const float4*)(src + i);
    dst[i + 0] = (__bf16)v.x;
    dst[i + 1] = (__bf16)v.y;
    dst[i + 2] = (__bf16)v.z;
    dst[i + 3] = (__bf16)v.w;
  }
}

// ---------------------------------------------------------------------------
// fp32 [K,N] weight -> bf16 [N,K] transpose (so B-fragments load contiguously)
// ---------------------------------------------------------------------------
__global__ void transpose_cvt(const float* __restrict__ W,
                              __bf16* __restrict__ Wt) {
  __shared__ __bf16 tile[32][33];
  const int n0 = blockIdx.x * 32, k0 = blockIdx.y * 32;
  const int tx = threadIdx.x, ty = threadIdx.y;  // 32 x 8 block
#pragma unroll
  for (int i = 0; i < 32; i += 8)
    tile[ty + i][tx] = (__bf16)W[(size_t)(k0 + ty + i) * E_ + n0 + tx];
  __syncthreads();
#pragma unroll
  for (int i = 0; i < 32; i += 8)
    Wt[(size_t)(n0 + ty + i) * E_ + k0 + tx] = tile[tx][ty + i];
}

// ---------------------------------------------------------------------------
// WMMA GEMM: C[M,N] = A[M,1024] * Wt[N,1024]^T + bias
// Block = 4 waves, block tile 64M x 64N. Wave w owns rows m0+w*16 and all
// 64 N columns (4 accumulators -> A-fragment reused 4x).
// Weight chunks are staged into a DOUBLE-BUFFERED LDS tile with
// global_load_async_to_lds_b128; the steady-state loop always stages chunk
// kc+1 before computing chunk kc (copy/compute overlap), and the LAST
// K-step is peeled so the loop body is branch-free: accumulators keep one
// phi placement and WMMA accumulates in place (no v_mov/v_nop hazards).
// mode 0: bf16 out, head-split [B,H,S,hd]   (Q, K)
// mode 1: bf16 out, transposed [B,H,hd,S]   (V)
// mode 2: f32  out, [M,E]                   (final projection -> d_out)
// ---------------------------------------------------------------------------
__global__ void gemm_bf16_kernel(const __bf16* __restrict__ A,
                                 const __bf16* __restrict__ Wt,
                                 const float* __restrict__ bias,
                                 __bf16* __restrict__ outB,
                                 float* __restrict__ outF, int mode) {
  __shared__ __bf16 bTile[2][64][40];  // ping-pong; 80B rows: conflict-free
  const int tid  = threadIdx.x;
  const int lane = tid & 31;
  const int w    = tid >> 5;
  const int lm   = lane & 15;
  const int half = lane >> 4;
  const int m0   = blockIdx.x * 64 + w * 16;
  const int n0   = blockIdx.y * 64;
  const int KC   = E_ / 32;  // 32 K-steps

  const __bf16* ar = A + (size_t)(m0 + lm) * E_;  // A: lane = row

  // async staging: thread t feeds LDS row t/2, k-half (t&1)*16 (32B = 2xb128)
  const __bf16* bsrc = Wt + (size_t)(n0 + (tid >> 1)) * E_ + (tid & 1) * 16;
  const unsigned bdst0 =
      (unsigned)(size_t)&bTile[0][tid >> 1][(tid & 1) * 16];  // LDS byte offset
  const unsigned bufStride = (unsigned)sizeof(bTile[0]);      // 5120 B

  // Issue one 64x32 bf16 chunk (4KB) as 2 async b128 per thread.
  // INST_OFFSET applies to both the global and LDS address (ISA 10.x).
  auto stage = [&](int chunk, int buf) {
    unsigned long long g = (unsigned long long)(bsrc + chunk * 32);
    unsigned d = bdst0 + (unsigned)buf * bufStride;
    asm volatile("global_load_async_to_lds_b128 %0, %1, off"
                 :: "v"(d), "v"(g) : "memory");
    asm volatile("global_load_async_to_lds_b128 %0, %1, off offset:16"
                 :: "v"(d), "v"(g) : "memory");
  };

  f32x8 acc[4];
#pragma unroll
  for (int n = 0; n < 4; ++n) acc[n] = (f32x8){};

  // one K-step of compute against LDS buffer `buf`
  auto compute = [&](int kc, int buf) {
    const __bf16* ap = ar + kc * 32;
    bf16x16 a = frag16(ap + half * 16, ap + half * 16 + 8);
    bf16x16 bfr[4];
#pragma unroll
    for (int n = 0; n < 4; ++n) {
      const __bf16* bp = &bTile[buf][n * 16 + lm][half * 16];
      bfr[n] = frag16(bp, bp + 8);
    }
#pragma unroll
    for (int n = 0; n < 4; ++n) acc[n] = wmma_bf16(a, bfr[n], acc[n]);
  };

  // prologue: stage chunk 0 into buffer 0
  stage(0, 0);
  asm volatile("s_wait_asynccnt 0x0" ::: "memory");
  __syncthreads();

  // steady state: branch-free body, always prefetching kc+1
#pragma unroll 1
  for (int kc = 0; kc < KC - 1; ++kc) {
    stage(kc + 1, (kc + 1) & 1);  // overlaps the compute below
    compute(kc, kc & 1);
    asm volatile("s_wait_asynccnt 0x0" ::: "memory");  // next chunk landed
    __syncthreads();                                   // block-wide visibility
  }
  compute(KC - 1, (KC - 1) & 1);  // peeled last step

  // ---- epilogue
  float bv[4];
#pragma unroll
  for (int n = 0; n < 4; ++n) bv[n] = bias[n0 + n * 16 + lm];
#pragma unroll
  for (int n = 0; n < 4; ++n) {
    const int nn = n0 + n * 16 + lm;
#pragma unroll
    for (int r = 0; r < 8; ++r) {
      const int m = m0 + half * 8 + r;  // C layout: row = half*8 + vgpr
      const float v = acc[n][r] + bv[n];
      if (mode == 2) {
        outF[(size_t)m * E_ + nn] = v;
      } else {
        const int bb = m >> 11;          // / S_
        const int s  = m & (S_ - 1);
        const int hh = nn >> 6;          // / HD_
        const int d  = nn & (HD_ - 1);
        const size_t off =
            (mode == 0) ? (((size_t)(bb * H_ + hh) * S_ + s) * HD_ + d)
                        : (((size_t)(bb * H_ + hh) * HD_ + d) * S_ + s);
        outB[off] = (__bf16)v;
      }
    }
  }
}

// ---------------------------------------------------------------------------
// Flash attention (causal), one 16-query tile per wave.
// Per 32-key chunk: 4 WMMA for Q*K^T, online softmax (shfl_xor within the
// 16-lane halves), P restaged via padded LDS into A-fragment layout
// (wave-local s_wait_dscnt), then 4 WMMA for P*V into 4 f32 accumulators.
// ---------------------------------------------------------------------------
__global__ void attn_kernel(const __bf16* __restrict__ Q,   // [B,H,S,hd]
                            const __bf16* __restrict__ K,   // [B,H,S,hd]
                            const __bf16* __restrict__ Vt,  // [B,H,hd,S]
                            __bf16* __restrict__ Y) {       // [B,S,E]
  __shared__ __bf16 pLds[4][16][40];   // 80B row stride: conflict-free b128
  const int lane = threadIdx.x & 31;
  const int w    = threadIdx.x >> 5;
  const int lm   = lane & 15;
  const int half = lane >> 4;
  const int bh   = blockIdx.y;  // b*H + h
  const int b    = bh >> 4;
  const int h    = bh & 15;
  const int q0   = (blockIdx.x * 4 + w) * 16;

  const __bf16* Qb = Q  + (size_t)bh * S_ * HD_;
  const __bf16* Kb = K  + (size_t)bh * S_ * HD_;
  const __bf16* Vb = Vt + (size_t)bh * HD_ * S_;

  // Q fragments (resident all loop): hd chunks [0,32) and [32,64)
  const __bf16* qr = Qb + (size_t)(q0 + lm) * HD_;
  const bf16x16 qa0 = frag16(qr + half * 16,      qr + half * 16 + 8);
  const bf16x16 qa1 = frag16(qr + 32 + half * 16, qr + 40 + half * 16);

  float m_i[8], l_i[8];
  f32x8 yacc[4];
#pragma unroll
  for (int r = 0; r < 8; ++r) { m_i[r] = -1e30f; l_i[r] = 0.f; }
#pragma unroll
  for (int n = 0; n < 4; ++n) yacc[n] = (f32x8){};

  const int nChunk = (q0 + 16 + 31) >> 5;  // causal: keys 0..q0+15
  for (int c = 0; c < nChunk; ++c) {
    const int c0 = c << 5;

    // ---- scores: two 16-key tiles, hd=64 -> 2 WMMA K-steps each
    f32x8 s[2];
#pragma unroll
    for (int t = 0; t < 2; ++t) {
      const __bf16* kr = Kb + (size_t)(c0 + t * 16 + lm) * HD_ + half * 16;
      bf16x16 kb0 = frag16(kr,      kr + 8);    // hd 0..31
      bf16x16 kb1 = frag16(kr + 32, kr + 40);   // hd 32..63
      f32x8 acc = (f32x8){};
      acc = wmma_bf16(qa0, kb0, acc);
      acc = wmma_bf16(qa1, kb1, acc);
      s[t] = acc;
    }

    // ---- scale (1/sqrt(64)) + causal mask
#pragma unroll
    for (int t = 0; t < 2; ++t) {
      const int key = c0 + t * 16 + lm;
#pragma unroll
      for (int r = 0; r < 8; ++r) {
        const int row = q0 + half * 8 + r;
        s[t][r] = (key <= row) ? s[t][r] * 0.125f : -1e30f;
      }
    }

    // ---- online softmax: row max/sum reductions across the 16-lane half
    float alpha[8];
#pragma unroll
    for (int r = 0; r < 8; ++r) {
      float v = fmaxf(s[0][r], s[1][r]);
#pragma unroll
      for (int off = 1; off < 16; off <<= 1)
        v = fmaxf(v, __shfl_xor(v, off, 32));
      const float nm = fmaxf(m_i[r], v);
      alpha[r] = __expf(m_i[r] - nm);
      m_i[r] = nm;
    }
#pragma unroll
    for (int r = 0; r < 8; ++r) {
      const float p0 = __expf(s[0][r] - m_i[r]);
      const float p1 = __expf(s[1][r] - m_i[r]);
      s[0][r] = p0;
      s[1][r] = p1;
      float rs = p0 + p1;
#pragma unroll
      for (int off = 1; off < 16; off <<= 1)
        rs += __shfl_xor(rs, off, 32);
      l_i[r] = l_i[r] * alpha[r] + rs;
    }
#pragma unroll
    for (int n = 0; n < 4; ++n)
#pragma unroll
      for (int r = 0; r < 8; ++r) yacc[n][r] *= alpha[r];

    // ---- C-layout P -> LDS (16x32 tile, padded rows), then A-fragment
#pragma unroll
    for (int t = 0; t < 2; ++t)
#pragma unroll
      for (int r = 0; r < 8; ++r)
        pLds[w][half * 8 + r][t * 16 + lm] = (__bf16)s[t][r];
    asm volatile("s_wait_dscnt 0x0" ::: "memory");  // wave-local LDS visibility
    const __bf16* pr = &pLds[w][lm][0];
    bf16x16 pa = frag16(pr + half * 16, pr + half * 16 + 8);

    // ---- P(16x32) @ V(32keys x 64hd): 4 WMMA, V^T loads are contiguous
#pragma unroll
    for (int n = 0; n < 4; ++n) {
      const __bf16* vr = Vb + (size_t)(n * 16 + lm) * S_ + c0 + half * 16;
      bf16x16 vb = frag16(vr, vr + 8);
      yacc[n] = wmma_bf16(pa, vb, yacc[n]);
    }
  }

  // ---- epilogue: divide by row sum, merge heads into [B,S,E] bf16
#pragma unroll
  for (int n = 0; n < 4; ++n) {
#pragma unroll
    for (int r = 0; r < 8; ++r) {
      const int row = q0 + half * 8 + r;
      const int col = h * HD_ + n * 16 + lm;
      const float v = yacc[n][r] / l_i[r];
      Y[((size_t)b * S_ + row) * E_ + col] = (__bf16)v;
    }
  }
}

// ---------------------------------------------------------------------------
// Host launcher
// ---------------------------------------------------------------------------
extern "C" void kernel_launch(void* const* d_in, const int* in_sizes, int n_in,
                              void* d_out, int out_size, void* d_ws,
                              size_t ws_size, hipStream_t stream) {
  (void)in_sizes; (void)n_in; (void)out_size; (void)ws_size;
  const float* x  = (const float*)d_in[0];
  const float* Wq = (const float*)d_in[1];
  const float* bq = (const float*)d_in[2];
  const float* Wk = (const float*)d_in[3];
  const float* bk = (const float*)d_in[4];
  const float* Wv = (const float*)d_in[5];
  const float* bv = (const float*)d_in[6];
  const float* Wp = (const float*)d_in[7];
  const float* bp = (const float*)d_in[8];

  char* ws = (char*)d_ws;
  size_t off = 0;
  auto take = [&](size_t elems) {
    __bf16* p = (__bf16*)(ws + off);
    off += elems * sizeof(__bf16);
    return p;
  };
  __bf16* xb  = take((size_t)M_ * E_);   // x in bf16
  __bf16* Wqt = take((size_t)E_ * E_);   // transposed bf16 weights
  __bf16* Wkt = take((size_t)E_ * E_);
  __bf16* Wvt = take((size_t)E_ * E_);
  __bf16* Wpt = take((size_t)E_ * E_);
  __bf16* Qh  = take((size_t)M_ * E_);   // [B,H,S,hd]
  __bf16* Kh  = take((size_t)M_ * E_);   // [B,H,S,hd]
  __bf16* Vth = take((size_t)M_ * E_);   // [B,H,hd,S]
  __bf16* Y   = take((size_t)M_ * E_);   // [B,S,E] attention output

  cvt_f32_bf16<<<(M_ * E_) / (256 * 4), 256, 0, stream>>>(x, xb, M_ * E_);

  dim3 tgrid(E_ / 32, E_ / 32), tblk(32, 8);
  transpose_cvt<<<tgrid, tblk, 0, stream>>>(Wq, Wqt);
  transpose_cvt<<<tgrid, tblk, 0, stream>>>(Wk, Wkt);
  transpose_cvt<<<tgrid, tblk, 0, stream>>>(Wv, Wvt);
  transpose_cvt<<<tgrid, tblk, 0, stream>>>(Wp, Wpt);

  dim3 ggrid(M_ / 64, E_ / 64);  // 64 x 16 blocks, 4 waves each
  gemm_bf16_kernel<<<ggrid, 128, 0, stream>>>(xb, Wqt, bq, Qh,  nullptr, 0);
  gemm_bf16_kernel<<<ggrid, 128, 0, stream>>>(xb, Wkt, bk, Kh,  nullptr, 0);
  gemm_bf16_kernel<<<ggrid, 128, 0, stream>>>(xb, Wvt, bv, Vth, nullptr, 1);

  attn_kernel<<<dim3(S_ / 64, B_ * H_), 128, 0, stream>>>(Qh, Kh, Vth, Y);

  gemm_bf16_kernel<<<ggrid, 128, 0, stream>>>(Y, Wpt, bp, nullptr,
                                              (float*)d_out, 2);
}
